// NeRTModel_60997125538302
// MI455X (gfx1250) — compile-verified, matrix operand order checked by reading, not run on hardware
//
#include <hip/hip_runtime.h>
#include <hip/hip_bf16.h>

// ---------------------------------------------------------------------------
// NeRT forward for MI455X (gfx1250, wave32, WMMA bf16 16x16x32 w/ f32 accum)
//
// Algebraic simplifications (exact up to fp rounding):
//  * word-attention pooling == mean over S  => w_q/w_k/b_q/b_k dead
//  * causal attention softmax is shift-invariant in the query term c_t and
//    b_attn => O(T^2) attention becomes an O(T) exp-weighted prefix scan
//  * LSTM x-projection hoisted out of the recurrence as one batched GEMM per
//    direction; only h@W_hh^T stays sequential (persistent 1-WG kernel/dir)
//
// LSTM critical-path optimizations:
//  * per-step x-gate slab (102KB) prefetched into LDS with double buffering
//    via GLOBAL_LOAD_ASYNC_TO_LDS_B128 + s_wait_asynccnt (gfx1250 async path)
//  * 32 waves per direction -> <=4 WMMA tiles per wave per step
// ---------------------------------------------------------------------------

typedef __bf16 bf16_t;
typedef __attribute__((ext_vector_type(16))) __bf16 v16bf;
typedef __attribute__((ext_vector_type(8)))  __bf16 v8bf;
typedef __attribute__((ext_vector_type(8)))  float  v8f;
typedef __attribute__((ext_vector_type(4)))  int    v4i;

#define B_   32
#define T_   256
#define S_   16
#define E_   400
#define H_   400
#define HC_  200
#define G4_  800     // 4*HC (i,f,g,o)
#define KE_  416     // E padded to multiple of 32
#define KH_  224     // HC padded to multiple of 32
#define NROWS_ (B_*T_)   // 8192

#if defined(__HIP_DEVICE_COMPILE__) && __has_builtin(__builtin_amdgcn_global_load_async_to_lds_b128)
#define LSTM_ASYNC 1
#else
#define LSTM_ASYNC 0
#endif

__device__ __forceinline__ float sigmoidf_(float x) { return 1.f / (1.f + __expf(-x)); }

#if LSTM_ASYNC
// Builtin signature (from hipcc diagnostics): param0 = int4 addrspace(1)*
// (global source), param1 = LDS dest, then imm offset + cpol.
__device__ __forceinline__ void async_b128(const void* gsrc, void* lds_dst) {
    __builtin_amdgcn_global_load_async_to_lds_b128(
        (__attribute__((address_space(1))) v4i*)gsrc,
        (__attribute__((address_space(3))) v4i*)lds_dst, 0, 0);
}
__device__ __forceinline__ void wait_async_le7() {
#if __has_builtin(__builtin_amdgcn_s_wait_asynccnt)
    __builtin_amdgcn_s_wait_asynccnt(7);
#else
    asm volatile("s_wait_asynccnt 0x7" ::: "memory");
#endif
}
__device__ __forceinline__ void wait_async_le0() {
#if __has_builtin(__builtin_amdgcn_s_wait_asynccnt)
    __builtin_amdgcn_s_wait_asynccnt(0);
#else
    asm volatile("s_wait_asynccnt 0x0" ::: "memory");
#endif
}
#endif

// Load one lane's 16 bf16 elements for a 16x32 WMMA operand.
// ISA 16-bit A layout: lane m (0..15) / hi = lane>>4; vector elems 0..7 hold
// K = 8*hi .. 8*hi+7, elems 8..15 hold K = 16+8*hi .. 16+8*hi+7.
__device__ __forceinline__ v16bf load_tile16(const bf16_t* p, int hi) {
    v8bf lo = *(const v8bf*)(p + 8 * hi);
    v8bf hh = *(const v8bf*)(p + 16 + 8 * hi);
    v16bf r;
#pragma unroll
    for (int i = 0; i < 8; i++) { r[i] = lo[i]; r[i + 8] = hh[i]; }
    return r;
}

// ---------------------------------------------------------------------------
// Generic C[M x N] = A[M x K](bf16) @ W[N x K]^T(bf16) + bias, one wave per
// 16x16 tile. K must be a multiple of 32 (we pad the staging buffers).
// flags: bit0 = store bf16 (else f32); bit1 = zero rows with t >= seq_lens[b]
//        (row index = b*T + t).
// ---------------------------------------------------------------------------
__global__ void gemm_bf16_wmma(const bf16_t* __restrict__ A, int lda,
                               const bf16_t* __restrict__ Bw, int ldb,
                               const float* __restrict__ bias,
                               void* __restrict__ Cout, int ldc,
                               int K, int flags, const int* __restrict__ seq_lens) {
    const int lane = threadIdx.x & 31;
    const int hi   = lane >> 4;
    const int mn   = lane & 15;
    const int row0 = blockIdx.y * 16;
    const int col0 = blockIdx.x * 16;

    const bf16_t* aRow = A  + (size_t)(row0 + mn) * lda;
    const bf16_t* bRow = Bw + (size_t)(col0 + mn) * ldb;

    v8f acc = {};
    for (int k = 0; k < K; k += 32) {
        v16bf av = load_tile16(aRow + k, hi);
        v16bf bv = load_tile16(bRow + k, hi);
        if (k + 32 < K) {                       // gfx1250 global_prefetch_b8
            __builtin_prefetch(aRow + k + 32, 0, 1);
            __builtin_prefetch(bRow + k + 32, 0, 1);
        }
        acc = __builtin_amdgcn_wmma_f32_16x16x32_bf16(false, av, false, bv,
                                                      (short)0, acc, false, false);
    }
    const float bb = bias ? bias[col0 + mn] : 0.f;
#pragma unroll
    for (int r = 0; r < 8; r++) {
        const int row = row0 + r + 8 * hi;      // C layout: VGPR r -> M = r + 8*hi
        float v = acc[r] + bb;
        if (flags & 2) {
            const int b = row >> 8, t = row & 255;
            if (t >= seq_lens[b]) v = 0.f;
        }
        if (flags & 1) ((bf16_t*)Cout)[(size_t)row * ldc + col0 + mn] = (bf16_t)v;
        else           ((float*) Cout)[(size_t)row * ldc + col0 + mn] = v;
    }
}

// ---------------------------------------------------------------------------
// Persistent BiLSTM recurrence: 1 workgroup per direction (blockIdx.x),
// 32 waves. Per step t:
//   1) issue async B128 loads of x-gate slab t+1 into the ping-pong LDS buf
//   2) s_wait_asynccnt <= 7  (own step-t loads complete; in-order) + barrier
//   3) G = h@W_hh^T via WMMA on LDS-resident bf16 h; epilogue RMW-adds into
//      the current x-gate LDS buffer (G += xgates)
//   4) gate nonlinearity; c update; h -> global f32 + LDS bf16 for step t+1
// LDS: 2 x 32*800 f32 (xgate ping-pong) + 32*200 f32 (c) + 32*224 bf16 (h)
//      = 239 KB  (< 320 KB/WGP)
// ---------------------------------------------------------------------------
#define XG_SLAB_ (32 * G4_)          // 25600 floats = 102400 B per step
#define NB128_   (XG_SLAB_ * 4 / 16) // 6400 B128 transfers per slab

__global__ __launch_bounds__(1024) void lstm_seq(
        const bf16_t* __restrict__ Whh_f, const bf16_t* __restrict__ Whh_b,
        const float* __restrict__ xg_f,   const float* __restrict__ xg_b,
        float* __restrict__ hbuf,   // [B,T,H]: fwd fills cols 0..HC-1
        float* __restrict__ hbr) {  // [B,T,HC]: bwd raw (pre-unreverse)
    extern __shared__ char smem_raw[];
    float*  xbuf0 = (float*)smem_raw;            // 32 x 800
    float*  xbuf1 = xbuf0 + XG_SLAB_;            // 32 x 800
    float*  cst   = xbuf1 + XG_SLAB_;            // 32 x 200
    bf16_t* hA    = (bf16_t*)(cst + 32 * HC_);   // 32 x 224 (cols 200.. zero)

    const int dir = blockIdx.x;
    const bf16_t* Whh = dir ? Whh_b : Whh_f;
    const float*  xg  = dir ? xg_b  : xg_f;

    const int tid  = threadIdx.x;
    const int wave = tid >> 5, lane = tid & 31;
    const int hi = lane >> 4, mn = lane & 15;
    const int nthreads = blockDim.x;             // 1024

    for (int i = tid; i < 32 * HC_; i += nthreads) cst[i] = 0.f;
    for (int i = tid; i < 32 * KH_; i += nthreads) hA[i] = (bf16_t)0.f;

    // --- prologue: stage x-gate slab for t = 0 into xbuf0 ---
#if LSTM_ASYNC
#pragma unroll
    for (int i = 0; i < 7; i++) {                // 7*1024 >= 6400, wrap dups ok
        int idx = tid + i * 1024; if (idx >= NB128_) idx -= NB128_;
        async_b128((const char*)xg + (size_t)idx * 16, (char*)xbuf0 + (size_t)idx * 16);
    }
    wait_async_le0();
#else
    for (int i = tid; i < XG_SLAB_; i += nthreads) xbuf0[i] = xg[i];
#endif
    __syncthreads();

    for (int t = 0; t < T_; ++t) {
        float* cur = (t & 1) ? xbuf1 : xbuf0;
        float* nxt = (t & 1) ? xbuf0 : xbuf1;

#if LSTM_ASYNC
        // 1) prefetch slab t+1 (safe: barrier above separates it from the
        //    previous step's readers of `nxt`)
        if (t + 1 < T_) {
            const char* src = (const char*)(xg + (size_t)(t + 1) * XG_SLAB_);
#pragma unroll
            for (int i = 0; i < 7; i++) {
                int idx = tid + i * 1024; if (idx >= NB128_) idx -= NB128_;
                async_b128(src + (size_t)idx * 16, (char*)nxt + (size_t)idx * 16);
            }
            wait_async_le7();   // exactly 7 just issued => older batch done
        } else {
            wait_async_le0();
        }
        __syncthreads();        // all waves' slab-t loads visible
#else
        (void)nxt;
#endif

        // 2) G = h @ Whh^T over 2 x 50 tiles of 16x16; epilogue adds into cur
        for (int tile = wave; tile < 100; tile += 32) {
            const int mt = tile & 1, nt = tile >> 1;
            const bf16_t* aRow = hA  + (size_t)(mt * 16 + mn) * KH_;
            const bf16_t* bRow = Whh + (size_t)(nt * 16 + mn) * KH_;
            v8f acc = {};
#pragma unroll
            for (int k = 0; k < KH_; k += 32) {
                v16bf av = load_tile16(aRow + k, hi);   // ds_load_b128
                v16bf bv = load_tile16(bRow + k, hi);
                acc = __builtin_amdgcn_wmma_f32_16x16x32_bf16(false, av, false, bv,
                                                              (short)0, acc, false, false);
            }
#pragma unroll
            for (int r = 0; r < 8; r++) {
                const int b = mt * 16 + r + 8 * hi;
                const int n = nt * 16 + mn;
                cur[b * G4_ + n] += acc[r];             // G + xgates, in LDS
            }
        }
        __syncthreads();

        // 3) gates -> (c, h)
        for (int idx = tid; idx < 32 * HC_; idx += nthreads) {
            const int b = idx / HC_, j = idx - b * HC_;
            const float gi = cur[b * G4_ + j];
            const float gf = cur[b * G4_ + HC_ + j];
            const float gg = cur[b * G4_ + 2 * HC_ + j];
            const float go = cur[b * G4_ + 3 * HC_ + j];
            const float c = sigmoidf_(gf) * cst[idx] + sigmoidf_(gi) * tanhf(gg);
            const float h = sigmoidf_(go) * tanhf(c);
            cst[idx] = c;
            hA[b * KH_ + j] = (bf16_t)h;
            if (dir == 0) hbuf[((size_t)b * T_ + t) * H_ + j]  = h;
            else          hbr [((size_t)b * T_ + t) * HC_ + j] = h;
        }
        __syncthreads();

#if !LSTM_ASYNC
        // synchronous staging of slab t+1 for the fallback path
        if (t + 1 < T_) {
            const float* src = xg + (size_t)(t + 1) * XG_SLAB_;
            for (int i = tid; i < XG_SLAB_; i += nthreads) nxt[i] = src[i];
            __syncthreads();
        }
#endif
    }
}

// f32 -> bf16 with zero K-padding (weights & activations staging)
__global__ void convert_pad(const float* __restrict__ src, bf16_t* __restrict__ dst,
                            int K, int Kp) {
    const int r = blockIdx.x;
    for (int k = threadIdx.x; k < Kp; k += blockDim.x)
        dst[(size_t)r * Kp + k] = (bf16_t)((k < K) ? src[(size_t)r * K + k] : 0.f);
}

__global__ void add_vec(const float* __restrict__ a, const float* __restrict__ b,
                        float* __restrict__ o, int n) {
    const int i = blockIdx.x * blockDim.x + threadIdx.x;
    if (i < n) o[i] = a[i] + b[i];
}

// pooled[n, :] = mean_s x2[n, s, :]  (word attention collapses to uniform mean)
__global__ void pool_mean(const float* __restrict__ x2, bf16_t* __restrict__ pooled) {
    const int n = blockIdx.x;      // 8192
    const int e = threadIdx.x;     // blockDim = KE_
    float acc = 0.f;
    if (e < E_) {
        const float* p = x2 + (size_t)n * S_ * E_ + e;
#pragma unroll
        for (int s = 0; s < S_; s++) acc += p[s * E_];
        acc *= (1.0f / S_);
    }
    pooled[(size_t)n * KE_ + e] = (bf16_t)acc;
}

// Stage x1 as bf16 rows (t*B+b) for the hoisted input projection:
// forward order, and per-length reversed order (rev = clamp(len-1-t, 0, T-1)).
__global__ void x_gather(const float* __restrict__ x1, const int* __restrict__ seq_lens,
                         bf16_t* __restrict__ xf, bf16_t* __restrict__ xb) {
    const int r = blockIdx.x;          // 8192 rows: r = t*32 + b
    const int t = r >> 5, b = r & 31;
    const int e = threadIdx.x;         // blockDim = KE_
    const int len = seq_lens[b];
    int tr = len - 1 - t; tr = tr < 0 ? 0 : (tr > T_ - 1 ? T_ - 1 : tr);
    float vf = 0.f, vb = 0.f;
    if (e < E_) {
        vf = x1[((size_t)b * T_ + t)  * E_ + e];
        vb = x1[((size_t)b * T_ + tr) * E_ + e];
    }
    xf[(size_t)r * KE_ + e] = (bf16_t)vf;
    xb[(size_t)r * KE_ + e] = (bf16_t)vb;
}

// Un-reverse backward h into hbuf[:, :, HC:], then a[b,t] = h[b,t,:].W_attn[:H]
__global__ void compose_a(const float* __restrict__ hbr, const int* __restrict__ seq_lens,
                          const float* __restrict__ Wattn, float* __restrict__ hbuf,
                          float* __restrict__ abuf) {
    __shared__ float red[128];
    const int r = blockIdx.x;          // 8192: r = b*T + t
    const int b = r >> 8, t = r & 255;
    const int len = seq_lens[b];
    int tr = len - 1 - t; tr = tr < 0 ? 0 : (tr > T_ - 1 ? T_ - 1 : tr);
    float* hrow = hbuf + (size_t)r * H_;
    const float* src = hbr + ((size_t)b * T_ + tr) * HC_;
    for (int j = threadIdx.x; j < HC_; j += blockDim.x) hrow[HC_ + j] = src[j];
    __syncthreads();
    float p = 0.f;
    for (int j = threadIdx.x; j < H_; j += blockDim.x) p += hrow[j] * Wattn[j];
    red[threadIdx.x] = p;
    __syncthreads();
    for (int s = 64; s > 0; s >>= 1) {
        if (threadIdx.x < s) red[threadIdx.x] += red[threadIdx.x + s];
        __syncthreads();
    }
    if (threadIdx.x == 0) abuf[r] = red[0];
}

// Causal attention as exp-weighted prefix scan; writes x1_att (bf16) into
// cat[:, 0:H] of the concat buffer for the final GEMM.
__global__ void causal_scan(const float* __restrict__ abuf, const float* __restrict__ hbuf,
                            bf16_t* __restrict__ cat) {
    __shared__ float ebuf[T_];
    __shared__ float dinv[T_];
    const int b = blockIdx.x;          // 32
    if (threadIdx.x < T_) ebuf[threadIdx.x] = abuf[b * T_ + threadIdx.x];
    __syncthreads();
    if (threadIdx.x == 0) {
        float m = -1e30f;
        for (int t = 0; t < T_; t++) m = fmaxf(m, ebuf[t]);
        float run = 0.f;
        for (int t = 0; t < T_; t++) {
            const float ev = __expf(ebuf[t] - m);
            ebuf[t] = ev; run += ev;
            dinv[t] = 1.f / run;
        }
    }
    __syncthreads();
    for (int j = threadIdx.x; j < H_; j += blockDim.x) {
        float num = 0.f;
        for (int t = 0; t < T_; t++) {
            num += ebuf[t] * hbuf[((size_t)b * T_ + t) * H_ + j];
            cat[((size_t)b * T_ + t) * (2 * H_) + j] = (bf16_t)(num * dinv[t]);
        }
    }
}

// ---------------------------------------------------------------------------
extern "C" void kernel_launch(void* const* d_in, const int* in_sizes, int n_in,
                              void* d_out, int out_size, void* d_ws, size_t ws_size,
                              hipStream_t stream) {
    (void)in_sizes; (void)n_in; (void)out_size; (void)ws_size;
    const float* x1        = (const float*)d_in[0];
    const float* x2        = (const float*)d_in[1];
    /* d_in[2] cate: unused by reference */
    const int*   seq_lens  = (const int*)  d_in[3];
    const float* w_ih_f    = (const float*)d_in[4];
    const float* w_hh_f    = (const float*)d_in[5];
    const float* b_ih_f    = (const float*)d_in[6];
    const float* b_hh_f    = (const float*)d_in[7];
    const float* w_ih_b    = (const float*)d_in[8];
    const float* w_hh_b    = (const float*)d_in[9];
    const float* b_ih_b    = (const float*)d_in[10];
    const float* b_hh_b    = (const float*)d_in[11];
    /* d_in[12..15] w_q,b_q,w_k,b_k: dead (pooling == mean over S) */
    const float* w_mlp_mha = (const float*)d_in[16];
    const float* b_mlp_mha = (const float*)d_in[17];
    const float* W_attn    = (const float*)d_in[18];
    /* d_in[19] b_attn: cancels in softmax */
    const float* w_mlp     = (const float*)d_in[20];
    const float* b_mlp     = (const float*)d_in[21];
    float* out = (float*)d_out;

    char* wsp = (char*)d_ws;
    auto bump = [&](size_t bytes) -> char* {
        char* p = wsp; wsp += (bytes + 255) & ~(size_t)255; return p;
    };
    bf16_t* Wmha_bf  = (bf16_t*)bump((size_t)H_  * KE_ * 2);
    bf16_t* Wihf_bf  = (bf16_t*)bump((size_t)G4_ * KE_ * 2);
    bf16_t* Wihb_bf  = (bf16_t*)bump((size_t)G4_ * KE_ * 2);
    bf16_t* Whhf_bf  = (bf16_t*)bump((size_t)G4_ * KH_ * 2);
    bf16_t* Whhb_bf  = (bf16_t*)bump((size_t)G4_ * KH_ * 2);
    bf16_t* Wmlp_bf  = (bf16_t*)bump((size_t)E_  * (2 * H_) * 2);
    float*  bsum_f   = (float*) bump(G4_ * 4);
    float*  bsum_b   = (float*) bump(G4_ * 4);
    bf16_t* pooled   = (bf16_t*)bump((size_t)NROWS_ * KE_ * 2);
    bf16_t* xf_bf    = (bf16_t*)bump((size_t)NROWS_ * KE_ * 2);
    bf16_t* xb_bf    = (bf16_t*)bump((size_t)NROWS_ * KE_ * 2);
    float*  xg_f     = (float*) bump((size_t)NROWS_ * G4_ * 4);
    float*  xg_b     = (float*) bump((size_t)NROWS_ * G4_ * 4);
    float*  hbuf     = (float*) bump((size_t)NROWS_ * H_ * 4);
    float*  hbr      = (float*) bump((size_t)NROWS_ * HC_ * 4);
    float*  abuf     = (float*) bump((size_t)NROWS_ * 4);
    bf16_t* catbuf   = (bf16_t*)bump((size_t)NROWS_ * (2 * H_) * 2);

    // --- stage weights/biases as padded bf16 ---
    convert_pad<<<H_,  256, 0, stream>>>(w_mlp_mha, Wmha_bf, E_,  KE_);
    convert_pad<<<G4_, 256, 0, stream>>>(w_ih_f,    Wihf_bf, E_,  KE_);
    convert_pad<<<G4_, 256, 0, stream>>>(w_ih_b,    Wihb_bf, E_,  KE_);
    convert_pad<<<G4_, 256, 0, stream>>>(w_hh_f,    Whhf_bf, HC_, KH_);
    convert_pad<<<G4_, 256, 0, stream>>>(w_hh_b,    Whhb_bf, HC_, KH_);
    convert_pad<<<E_,  256, 0, stream>>>(w_mlp,     Wmlp_bf, 2 * H_, 2 * H_);
    add_vec<<<(G4_ + 255) / 256, 256, 0, stream>>>(b_ih_f, b_hh_f, bsum_f, G4_);
    add_vec<<<(G4_ + 255) / 256, 256, 0, stream>>>(b_ih_b, b_hh_b, bsum_b, G4_);

    // --- activations staging ---
    pool_mean<<<NROWS_, KE_, 0, stream>>>(x2, pooled);
    x_gather <<<NROWS_, KE_, 0, stream>>>(x1, seq_lens, xf_bf, xb_bf);

    // --- hoisted LSTM input projections: [8192x800] = x @ W_ih^T + (b_ih+b_hh)
    gemm_bf16_wmma<<<dim3(G4_ / 16, NROWS_ / 16), 32, 0, stream>>>(
        xf_bf, KE_, Wihf_bf, KE_, bsum_f, xg_f, G4_, KE_, 0, nullptr);
    gemm_bf16_wmma<<<dim3(G4_ / 16, NROWS_ / 16), 32, 0, stream>>>(
        xb_bf, KE_, Wihb_bf, KE_, bsum_b, xg_b, G4_, KE_, 0, nullptr);

    // --- x2p MLP, bf16 directly into cat[:, H:2H] ---
    gemm_bf16_wmma<<<dim3(H_ / 16, NROWS_ / 16), 32, 0, stream>>>(
        pooled, KE_, Wmha_bf, KE_, b_mlp_mha, (void*)(catbuf + H_), 2 * H_,
        KE_, /*bf16 out*/1, nullptr);

    // --- persistent BiLSTM recurrence (1 WG per direction, 32 waves) ---
    const size_t lstm_smem = (size_t)2 * XG_SLAB_ * 4      // xgate ping-pong
                           + (size_t)32 * HC_ * 4          // c state
                           + (size_t)32 * KH_ * 2;         // bf16 h  (~239 KB)
    lstm_seq<<<2, 1024, lstm_smem, stream>>>(Whhf_bf, Whhb_bf, xg_f, xg_b, hbuf, hbr);

    // --- un-reverse backward half + attention key scores ---
    compose_a<<<NROWS_, 128, 0, stream>>>(hbr, seq_lens, W_attn, hbuf, abuf);

    // --- causal attention as prefix scan -> cat[:, 0:H] ---
    causal_scan<<<B_, 512, 0, stream>>>(abuf, hbuf, catbuf);

    // --- final MLP with ragged-length masking -> d_out [B,T,E] f32 ---
    gemm_bf16_wmma<<<dim3(E_ / 16, NROWS_ / 16), 32, 0, stream>>>(
        catbuf, 2 * H_, Wmlp_bf, 2 * H_, b_mlp, out, E_,
        2 * H_, /*mask*/2, seq_lens);
}
